// PINGNet_14559939134166
// MI455X (gfx1250) — matrix operand order repaired and checked by reading, not compiled, not run on hardware
//
#include <hip/hip_runtime.h>

typedef __attribute__((ext_vector_type(16))) _Float16 v16h;
typedef __attribute__((ext_vector_type(8)))  _Float16 v8h;
typedef __attribute__((ext_vector_type(8)))  float    v8f;

// Model constants (forward pass only; surrogate-grad parts are identity forward)
#define K_E_L     (-65.0f)
#define K_E_I     (-80.0f)
#define K_V_TH    (-50.0f)
#define K_V_RESET (-65.0f)
#define K_V_FLOOR (-200.0f)
#define DECAY_AMPA 0.8824968934059143f   // exp(-0.25/2)
#define DECAY_GABA 0.9726044535636902f   // exp(-0.25/9)

// ---- WMMA helpers -----------------------------------------------------------

static __device__ inline v8f wmma16(v16h a, v16h b, v8f c) {
  // D = A(16x32 f16) * B(32x16 f16) + C(16x16 f32)
  return __builtin_amdgcn_wmma_f32_16x16x32_f16(
      /*neg_a=*/false, a, /*neg_b=*/false, b,
      /*c_mod=*/(short)0, c, /*reuse_a=*/false, /*reuse_b=*/false);
}

// Gather one lane's A fragment from a row-major f16 matrix in LDS.
// 16-bit A 16x32 layout: lanes 0-15 hold (M=lane, K=koff+0..7, koff+16..23),
// lanes 16-31 hold (M=lane-16, K=koff+8..15, koff+24..31).
static __device__ inline v16h load_afrag(const _Float16* mat, int ldm, int koff, int lane) {
  const int r  = lane & 15;
  const int bk = koff + ((lane >> 4) << 3);
  const _Float16* p = mat + r * ldm + bk;
  union { v16h v; v8h h[2]; } u;
  u.h[0] = *(const v8h*)(p);
  u.h[1] = *(const v8h*)(p + 16);
  return u.v;
}

// Pre-packed B fragment: 32 contiguous bytes per lane.
static __device__ inline v16h load_bfrag(const _Float16* frags, int tile, int lane) {
  return *(const v16h*)(frags + (tile << 9) + (lane << 4));
}

// Pack a row-major f32 weight [Kdim x Nreal] into per-tile WMMA B fragments.
// B 32x16 layout: lane l holds column N = nt*16 + (l&15); its 16 f16 are
// K = kt*32 + (l>=16 ? 8 : 0) + (e&7) + (e>=8 ? 16 : 0), e=0..15. Columns beyond Nreal are zero-padded.
static __device__ inline void pack_bfrags(_Float16* dst, const float* W,
                                          int nKT, int nNT, int Nreal,
                                          bool clampPos, int tid, int nthr) {
  const int total = nKT * nNT * 512;
  for (int i = tid; i < total; i += nthr) {
    const int tile = i >> 9;
    const int rem  = i & 511;
    const int lane = rem >> 4;
    const int e    = rem & 15;
    const int kt = tile / nNT, nt = tile % nNT;
    const int k = kt * 32 + ((lane >= 16) ? 8 : 0) + (e & 7) + ((e >> 3) << 4);
    const int n = nt * 16 + (lane & 15);
    float v = 0.0f;
    if (n < Nreal) {
      v = W[k * Nreal + n];
      if (clampPos) v = fmaxf(v, 0.0f);   // Dale's law clamp on feedforward weights
    }
    dst[i] = (_Float16)v;
  }
}

// ---- LIF neuron (forward) ---------------------------------------------------

static __device__ inline float lif(float& v, int& ref, float I,
                                   float dtOverC, float gL, int refSteps) {
  const float dv = dtOverC * (-gL * (v - K_E_L) + I);
  v = fmaxf(v + dv, K_V_FLOOR);
  ref = (ref > 0) ? (ref - 1) : 0;
  const bool can = (ref == 0);
  const bool s   = can && (v >= K_V_TH);
  if (s || !can) v = K_V_RESET;
  if (s) ref = refSteps;
  return s ? 1.0f : 0.0f;
}

// ---- Persistent PING kernel: one workgroup per 16 batch rows ----------------
// Block = 512 threads = 16 waves. Wave w owns E columns [16w, 16w+16).
// Waves 0-3 additionally own the I population; wave 4 owns the output layer.

__global__ void __launch_bounds__(512)
ping_persistent(const float* __restrict__ inp,   // [T,64,64] input spikes
                const float* __restrict__ Wff0,  // [64,256]
                const float* __restrict__ Wff1,  // [256,10]
                const float* __restrict__ Wee,   // [256,256]
                const float* __restrict__ Wei,   // [256,64]
                const float* __restrict__ Wie,   // [64,256]
                float* __restrict__ out,         // [64,10] spike counts
                int T) {
  // Weight fragments (packed once): 232 KB
  __shared__ alignas(32) _Float16 fragW0 [ 32 * 512];   // kt(2)  x nt(16)
  __shared__ alignas(32) _Float16 fragWee[128 * 512];   // kt(8)  x nt(16)
  __shared__ alignas(32) _Float16 fragWie[ 32 * 512];   // kt(2)  x nt(16)
  __shared__ alignas(32) _Float16 fragWei[ 32 * 512];   // kt(8)  x nt(4)
  __shared__ alignas(32) _Float16 fragW1 [  8 * 512];   // kt(8)  x nt(1), N padded 10->16
  // Per-step spike matrices (row-major f16): 12 KB
  __shared__ alignas(32) _Float16 sE [16 * 256];
  __shared__ alignas(32) _Float16 sI [16 * 64];
  __shared__ alignas(32) _Float16 sIn[16 * 64];

  const int tid  = threadIdx.x;
  const int lane = tid & 31;
  const int w    = tid >> 5;          // wave id 0..15 (wave32)
  const int r0   = blockIdx.x << 4;   // first batch row of this WG

  // ---- one-time setup: pack weights to WMMA B layout, zero spike state ----
  pack_bfrags(fragW0,  Wff0, 2, 16, 256, true,  tid, 512);
  pack_bfrags(fragWee, Wee,  8, 16, 256, false, tid, 512);
  pack_bfrags(fragWie, Wie,  2, 16, 256, false, tid, 512);
  pack_bfrags(fragWei, Wei,  8,  4,  64, false, tid, 512);
  pack_bfrags(fragW1,  Wff1, 8,  1,  10, true,  tid, 512);
  for (int i = tid; i < 16 * 256; i += 512) sE[i] = (_Float16)0.0f;
  for (int i = tid; i < 16 * 64;  i += 512) sI[i] = (_Float16)0.0f;

  // ---- register-resident neuron state, matching WMMA C/D layout ----
  // element j <-> (M = j + 8*(lane>=16), N = lane&15), column = 16*w + N
  float vE[8], geE[8], giE[8]; int rE[8];
  float vI[8], geI[8];         int rI[8];
  float vO[8], gO[8], accO[8]; int rO[8];
#pragma unroll
  for (int j = 0; j < 8; ++j) {
    vE[j] = K_E_L; geE[j] = 0.f; giE[j] = 0.f; rE[j] = 0;
    vI[j] = K_E_L; geI[j] = 0.f;               rI[j] = 0;
    vO[j] = K_E_L; gO[j]  = 0.f; accO[j] = 0.f; rO[j] = 0;
  }

  const int Mrow = ((lane >> 4) << 3);        // +0 or +8
  const int nloc = lane & 15;
  const float* slice0 = inp + (size_t)r0 * 64; // rows r0..r0+15 are contiguous

  __syncthreads();

  // ---- 4000-step scan ----
  for (int t = 0; t < T; ++t) {
    // stage input spikes for this WG's batch rows (f32 -> f16), prefetch next
    {
      const float* base = slice0 + (size_t)t * 4096;
      sIn[tid]       = (_Float16)base[tid];
      sIn[tid + 512] = (_Float16)base[tid + 512];
      if (t + 1 < T) __builtin_prefetch(base + 4096 + tid);
    }
    __syncthreads();   // sIn ready; previous-step sE/sI stores visible

    // --- GEMM phase (uses previous-step spikes) ---
    v8f cE = {0.f,0.f,0.f,0.f,0.f,0.f,0.f,0.f};
#pragma unroll
    for (int kt = 0; kt < 2; ++kt)   // s_in @ W0
      cE = wmma16(load_afrag(sIn, 64, kt * 32, lane),
                  load_bfrag(fragW0, kt * 16 + w, lane), cE);
#pragma unroll
    for (int kt = 0; kt < 8; ++kt)   // s_e @ W_ee
      cE = wmma16(load_afrag(sE, 256, kt * 32, lane),
                  load_bfrag(fragWee, kt * 16 + w, lane), cE);

    v8f cG = {0.f,0.f,0.f,0.f,0.f,0.f,0.f,0.f};
#pragma unroll
    for (int kt = 0; kt < 2; ++kt)   // s_i @ W_ie
      cG = wmma16(load_afrag(sI, 64, kt * 32, lane),
                  load_bfrag(fragWie, kt * 16 + w, lane), cG);

    v8f cI = {0.f,0.f,0.f,0.f,0.f,0.f,0.f,0.f};
    if (w < 4) {
#pragma unroll
      for (int kt = 0; kt < 8; ++kt) // s_e @ W_ei
        cI = wmma16(load_afrag(sE, 256, kt * 32, lane),
                    load_bfrag(fragWei, kt * 4 + w, lane), cI);
    }
    __syncthreads();   // all GEMM reads of sE/sI done before overwrite

    // --- LIF phase: E population (all waves), I population (waves 0-3) ---
#pragma unroll
    for (int j = 0; j < 8; ++j) {
      geE[j] = (geE[j] + cE[j]) * DECAY_AMPA;
      giE[j] = (giE[j] + cG[j]) * DECAY_GABA;
      const float Icur = geE[j] * (0.0f - vE[j]) + giE[j] * (K_E_I - vE[j]);
      const float s = lif(vE[j], rE[j], Icur, /*dt/C=*/0.25f, /*gL=*/0.05f, 12);
      sE[(j + Mrow) * 256 + (w << 4) + nloc] = (_Float16)s;
    }
    if (w < 4) {
#pragma unroll
      for (int j = 0; j < 8; ++j) {
        geI[j] = (geI[j] + cI[j]) * DECAY_AMPA;
        const float Icur = geI[j] * (0.0f - vI[j]);
        const float s = lif(vI[j], rI[j], Icur, /*dt/C=*/0.5f, /*gL=*/0.1f, 6);
        sI[(j + Mrow) * 64 + (w << 4) + nloc] = (_Float16)s;
      }
    }
    __syncthreads();   // fresh sE visible

    // --- output layer (wave 4), driven by fresh E spikes ---
    if (w == 4) {
      v8f cO = {0.f,0.f,0.f,0.f,0.f,0.f,0.f,0.f};
#pragma unroll
      for (int kt = 0; kt < 8; ++kt)  // s_e(new) @ W1
        cO = wmma16(load_afrag(sE, 256, kt * 32, lane),
                    load_bfrag(fragW1, kt, lane), cO);
#pragma unroll
      for (int j = 0; j < 8; ++j) {
        gO[j] = (gO[j] + cO[j]) * DECAY_AMPA;
        const float Icur = gO[j] * (0.0f - vO[j]);
        accO[j] += lif(vO[j], rO[j], Icur, 0.25f, 0.05f, 12);
      }
    }
  }

  // ---- emit spike counts [64,10] ----
  if (w == 4) {
#pragma unroll
    for (int j = 0; j < 8; ++j) {
      if (nloc < 10) out[(r0 + j + Mrow) * 10 + nloc] = accO[j];
    }
  }
}

extern "C" void kernel_launch(void* const* d_in, const int* in_sizes, int n_in,
                              void* d_out, int out_size, void* d_ws, size_t ws_size,
                              hipStream_t stream) {
  (void)n_in; (void)out_size; (void)d_ws; (void)ws_size;
  const float* inp  = (const float*)d_in[0];  // input_spikes [T,64,64]
  const float* Wff0 = (const float*)d_in[1];  // [64,256]
  const float* Wff1 = (const float*)d_in[2];  // [256,10]
  const float* Wee  = (const float*)d_in[3];  // [256,256]
  const float* Wei  = (const float*)d_in[4];  // [256,64]
  const float* Wie  = (const float*)d_in[5];  // [64,256]
  float* out = (float*)d_out;
  const int T = in_sizes[0] / (64 * 64);
  // 4 workgroups (one per 16 batch rows), 16 waves each; persistent over T steps.
  ping_persistent<<<4, 512, 0, stream>>>(inp, Wff0, Wff1, Wee, Wei, Wie, out, T);
}
